// VNETDetector_3942779978170
// MI455X (gfx1250) — matrix operand order repaired and checked by reading, not compiled, not run on hardware
//
#include <hip/hip_runtime.h>
#include <hip/hip_bf16.h>

typedef __attribute__((ext_vector_type(16))) _Float16 v16h;
typedef __attribute__((ext_vector_type(8)))  float    v8f;

#define NSTATES 16
#define HIDDEN  100
#define CHUNK   512
#define SROWS   16            // llr rows staged per async buffer (CHUNK % SROWS == 0)
#define BIG     1.0e30f

// gfx1250 async global->LDS path (guarded: falls back to direct vector loads)
#if defined(__has_builtin)
#if __has_builtin(__builtin_amdgcn_global_load_async_to_lds_b128) && \
    __has_builtin(__builtin_amdgcn_s_wait_asynccnt)
#define HAVE_ASYNC_LDS 1
#endif
#endif
#ifndef HAVE_ASYNC_LDS
#define HAVE_ASYNC_LDS 0
#endif

#if HAVE_ASYNC_LDS
// builtin prototype (from hipcc diagnostic): pointee is a GCC vector of 4 ints,
// src in global (AS1), dst in LDS (AS3), plus two immediate ints (offset, cpol).
typedef int v4i_gcc __attribute__((vector_size(4 * sizeof(int))));
typedef __attribute__((address_space(1))) v4i_gcc as1_v4i;
typedef __attribute__((address_space(3))) v4i_gcc as3_v4i;
#endif

// =====================================================================
// Kernel 1: fused  h = relu(rx*W1+b1) -> logits = h@W2^T+b2  via WMMA,
// then log-softmax / llr / confident-bit / confidence, all in registers.
// Each wave owns a 16-timestep tile: 4x v_wmma_f32_16x16x32_f16 (K 100->128).
// =====================================================================
__global__ __launch_bounds__(256) void vnet_logits_wmma(
    const float* __restrict__ rx, const float* __restrict__ W1,
    const float* __restrict__ b1, const float* __restrict__ W2,
    const float* __restrict__ b2, float* __restrict__ llr,
    float* __restrict__ out, int T)
{
  const int lane = threadIdx.x & 31;
  const int wave = threadIdx.x >> 5;
  const int hi   = lane >> 4;          // 16-lane half select
  const int n    = lane & 15;          // state column for B/C/D; row M for A
  const int tile = blockIdx.x * 8 + wave;
  const int t0   = tile * 16;

  // ---- B fragments: B[k][n] = W2[n][k]; contiguous k order used here:
  // lanes 0-15 hold K = 0..15 of each 32-chunk, lanes 16-31 hold K = 16..31.
  v16h bf[4];
  #pragma unroll
  for (int c = 0; c < 4; ++c) {
    #pragma unroll
    for (int j = 0; j < 16; ++j) {
      const int k = c * 32 + hi * 16 + j;
      bf[c][j] = (_Float16)((k < HIDDEN) ? W2[n * HIDDEN + k] : 0.0f);
    }
  }

  // ---- A fragments: A[m][k] = relu(rx[t0+m]*W1[k]+b1[k]); contiguous k
  // order, as B: lane m holds row m; half j -> K = hi*16 + j.
  const int  tA = t0 + n;
  const float x = (tA < T) ? rx[tA] : 0.0f;

  v8f acc;
  const float bb = b2[n];
  #pragma unroll
  for (int r = 0; r < 8; ++r) acc[r] = bb;     // C = broadcast bias

  #pragma unroll
  for (int c = 0; c < 4; ++c) {
    v16h af;
    #pragma unroll
    for (int j = 0; j < 16; ++j) {
      const int k = c * 32 + hi * 16 + j;
      float h = 0.0f;
      if (k < HIDDEN) {
        h = __builtin_fmaf(x, W1[k], b1[k]);
        h = h > 0.0f ? h : 0.0f;
      }
      af[j] = (_Float16)h;
    }
    acc = __builtin_amdgcn_wmma_f32_16x16x32_f16(false, af, false, bf[c],
                                                 (short)0, acc, false, false);
  }

  // ---- D layout: lane holds logits[row = r+8*hi][state = n], r = 0..7.
  #pragma unroll
  for (int r = 0; r < 8; ++r) {
    const int   t  = t0 + r + 8 * hi;
    const float lg = acc[r];

    float mv = lg; int mi = n;                         // max + first-argmax
    #pragma unroll
    for (int s = 8; s >= 1; s >>= 1) {
      const float ov = __shfl_xor(mv, s);
      const int   oi = __shfl_xor(mi, s);
      if (ov > mv || (ov == mv && oi < mi)) { mv = ov; mi = oi; }
    }
    float sum = __expf(lg - mv);                       // sum of exp
    #pragma unroll
    for (int s = 8; s >= 1; s >>= 1) sum += __shfl_xor(sum, s);
    const float ls = __logf(sum);

    if (t < T) {
      llr[t * NSTATES + n] = mv + ls - lg;             // llr = -log_softmax
      if (n == 0) {
        out[T + t]     = (float)(mi & 1);              // confident_bits
        out[2 * T + t] = 1.0f / sum;                   // confidence_word
      }
    }
  }
}

// ---- one ACS step on a column vector held in registers -----------------
__device__ __forceinline__ void acs_step(float v[16], const float4 l0,
                                         const float4 l1, const float4 l2,
                                         const float4 l3)
{
  const float lr[16] = { l0.x, l0.y, l0.z, l0.w, l1.x, l1.y, l1.z, l1.w,
                         l2.x, l2.y, l2.z, l2.w, l3.x, l3.y, l3.z, l3.w };
  float p[16];
  #pragma unroll
  for (int q = 0; q < 16; ++q) p[q] = v[q] + lr[q];
  #pragma unroll
  for (int i = 0; i < 16; ++i)
    v[i] = fminf(p[(2 * i) & 15], p[(2 * i + 1) & 15]);
}

// =====================================================================
// Kernel 2: per-chunk 16x16 min-plus transition matrix.
// Lane s of each 16-lane group carries column s. Each step consumes one
// 64B llr row (broadcast across the group). With async-to-LDS available,
// rows are double-buffered into LDS by global_load_async_to_lds_b128 and
// consumed via broadcast ds_load_b128 behind s_wait_asynccnt.
// =====================================================================
__global__ __launch_bounds__(256) void vnet_chunk_mats(
    const float* __restrict__ llr, float* __restrict__ cmat, int T, int nChunks)
{
#if HAVE_ASYNC_LDS
  __shared__ float stage[16][2][SROWS * NSTATES];    // 2KB/group, 32KB/block
#endif
  const int grp = threadIdx.x >> 4;
  const int s   = threadIdx.x & 15;
  const int c   = blockIdx.x * 16 + grp;
  if (c >= nChunks) return;

  float v[16];
  #pragma unroll
  for (int i = 0; i < 16; ++i) v[i] = (i == s) ? 0.0f : BIG;

  const int t0c  = c * CHUNK;
  const int tEnd = min(t0c + CHUNK, T);

#if HAVE_ASYNC_LDS
  // stage SROWS rows (SROWS/4 async b128 ops: 16 lanes x 16B = 4 rows each)
  auto issue_fill = [&](int buf, int tBase) {
    #pragma unroll
    for (int q = 0; q < SROWS / 4; ++q) {
      const float* g = llr + (size_t)(tBase + q * 4) * NSTATES + s * 4;
      float*       l = &stage[grp][buf][q * 4 * NSTATES + s * 4];
      __builtin_amdgcn_global_load_async_to_lds_b128((as1_v4i*)g, (as3_v4i*)l,
                                                     0, 0);
    }
  };
  issue_fill(0, t0c);
  int buf = 0;
  for (int tb = t0c; tb < tEnd; tb += SROWS) {
    issue_fill(buf ^ 1, tb + SROWS);                 // prefetch next buffer
    __builtin_amdgcn_s_wait_asynccnt(SROWS / 4);     // current buffer landed
    asm volatile("" ::: "memory");
    const int nr = min(SROWS, tEnd - tb);
    for (int r = 0; r < nr; ++r) {
      const float4* row = (const float4*)&stage[grp][buf][r * NSTATES];
      acs_step(v, row[0], row[1], row[2], row[3]);   // broadcast ds_load_b128
    }
    buf ^= 1;
  }
  __builtin_amdgcn_s_wait_asynccnt(0);
#else
  for (int t = t0c; t < tEnd; ++t) {
    const float4* row = (const float4*)(llr + (size_t)t * NSTATES);
    acs_step(v, row[0], row[1], row[2], row[3]);     // group-shared 64B row
  }
#endif

  #pragma unroll
  for (int i = 0; i < 16; ++i) cmat[(c * 16 + i) * 16 + s] = v[i];
}

// =====================================================================
// Kernel 3: sequential exclusive scan of chunk matrices (~489 tiny steps).
// pref[c] = state vector at the start of chunk c (pref[0] = zeros).
// =====================================================================
__global__ void vnet_chunk_scan(const float* __restrict__ cmat,
                                float* __restrict__ pref, int nChunks)
{
  const int i = threadIdx.x;
  if (i >= 16) return;
  float v[16];
  #pragma unroll
  for (int s = 0; s < 16; ++s) v[s] = 0.0f;

  for (int c = 0; c < nChunks; ++c) {
    pref[c * 16 + i] = v[i];
    float nv = BIG;
    #pragma unroll
    for (int s = 0; s < 16; ++s)
      nv = fminf(nv, cmat[(c * 16 + i) * 16 + s] + v[s]);
    #pragma unroll
    for (int s = 0; s < 16; ++s) v[s] = __shfl(nv, s, 16);
  }
}

// =====================================================================
// Kernel 4: replay. One lane per chunk: start from pref[c], re-run ACS,
// emit bit = argmin(in_prob) % 2 BEFORE each update (matches scan order).
// =====================================================================
__global__ __launch_bounds__(256) void vnet_emit(
    const float* __restrict__ llr, const float* __restrict__ pref,
    float* __restrict__ out, int T, int nChunks)
{
  const int c = blockIdx.x * blockDim.x + threadIdx.x;
  if (c >= nChunks) return;

  float v[16];
  #pragma unroll
  for (int i = 0; i < 16; ++i) v[i] = pref[c * 16 + i];

  const int tEnd = min((c + 1) * CHUNK, T);
  for (int t = c * CHUNK; t < tEnd; ++t) {
    float bv = v[0]; int bi = 0;                       // first-min argmin
    #pragma unroll
    for (int i = 1; i < 16; ++i) if (v[i] < bv) { bv = v[i]; bi = i; }
    out[t] = (float)(bi & 1);

    __builtin_prefetch(llr + (t + 1) * NSTATES, 0, 0); // global_prefetch_b8
    const float4* row = (const float4*)(llr + (size_t)t * NSTATES);
    acs_step(v, row[0], row[1], row[2], row[3]);
  }
}

extern "C" void kernel_launch(void* const* d_in, const int* in_sizes, int n_in,
                              void* d_out, int out_size, void* d_ws, size_t ws_size,
                              hipStream_t stream) {
  const float* rx = (const float*)d_in[0];
  const float* W1 = (const float*)d_in[1];
  const float* b1 = (const float*)d_in[2];
  const float* W2 = (const float*)d_in[3];
  const float* b2 = (const float*)d_in[4];
  float* out = (float*)d_out;

  const int T       = in_sizes[0];                 // rx is [T,1]
  const int nChunks = (T + CHUNK - 1) / CHUNK;

  // workspace: llr [T*16] f32 | chunk mats [nChunks*256] | prefix [nChunks*16]
  float* llr  = (float*)d_ws;
  float* cmat = llr + (size_t)T * NSTATES;
  float* pref = cmat + (size_t)nChunks * NSTATES * NSTATES;

  const int tiles   = (T + 15) / 16;
  const int blocks1 = (tiles + 7) / 8;             // 8 waves (tiles) per block

  vnet_logits_wmma<<<blocks1, 256, 0, stream>>>(rx, W1, b1, W2, b2, llr, out, T);
  vnet_chunk_mats<<<(nChunks + 15) / 16, 256, 0, stream>>>(llr, cmat, T, nChunks);
  vnet_chunk_scan<<<1, 32, 0, stream>>>(cmat, pref, nChunks);
  vnet_emit<<<(nChunks + 255) / 256, 256, 0, stream>>>(llr, pref, out, T, nChunks);
}
